// MolecularE3nnTransformerUpdate_32186484916933
// MI455X (gfx1250) — matrix operand order repaired and checked by reading, not compiled, not run on hardware
//
#include <hip/hip_runtime.h>
#include <hip/hip_bf16.h>
#include <math.h>

// ---------------------------------------------------------------------------
// MolecularE3nnTransformerUpdate for MI455X (gfx1250, wave32)
//
// The two [N,256]@[256,256] ReLU layers (65.5 GFLOP) run on
// v_wmma_f32_16x16x32_f16 with: weights pre-converted to f16 in B-fragment
// order (contiguous v16h per lane/k-step), activations kept f16 row-major
// (A fragment = two contiguous v8h loads), and 5 row tiles per wave sharing
// each B fragment. Equivariant 11x11x{1,3,5} ops + segment sums stay scalar
// VALU + atomics (~2 GFLOP, trivial vs 23.3 TB/s).
// ---------------------------------------------------------------------------

typedef __attribute__((ext_vector_type(16))) _Float16 v16h;
typedef __attribute__((ext_vector_type(8)))  _Float16 v8h;
typedef __attribute__((ext_vector_type(8)))  float    v8f;

#define SQ3f  1.7320508075688772f
#define C15f  3.8729833462074170f
#define C5f   2.2360679774997896f
#define INV_SQ3f   0.5773502691896258f
#define INV_SQ5f   0.4472135954999579f
#define INV_SQ363f 0.052486388108147424f   // 1/sqrt(363)
#define INV_SQ11f  0.30151134457776363f    // 1/sqrt(11)
#define EPS2f 1e-10f                        // EPS^2
#define MT 5                                // row tiles per wave in GEMM

__device__ __forceinline__ float sigm(float t) { return 1.0f / (1.0f + __expf(-t)); }

// ---------------------------------------------------------------------------
// K0: h = x @ embd_w + embd_b ; spherical harmonics (lmax=2, component norm)
// Pin layout per node (stride 19): [0..9]=h, [10]=sh0(=1), [11..13]=sh1..3,
// [14..18]=sh4..8
// ---------------------------------------------------------------------------
__global__ void k_embed(const float* __restrict__ x, const float* __restrict__ pos,
                        const float* __restrict__ ew, const float* __restrict__ eb,
                        float* __restrict__ Pin, int n) {
  int node = blockIdx.x * blockDim.x + threadIdx.x;
  if (node >= n) return;
  const float* xv = x + (size_t)node * 10;
  float* o = Pin + (size_t)node * 19;
  for (int i = 0; i < 10; ++i) {
    float a = eb[i];
    for (int c = 0; c < 10; ++c) a += xv[c] * ew[c * 10 + i];
    o[i] = a;
  }
  float px = pos[(size_t)node * 3 + 0];
  float py = pos[(size_t)node * 3 + 1];
  float pz = pos[(size_t)node * 3 + 2];
  float inv = rsqrtf(px * px + py * py + pz * pz);
  float X = px * inv, Y = py * inv, Z = pz * inv;
  o[10] = 1.0f;
  o[11] = SQ3f * X;  o[12] = SQ3f * Y;  o[13] = SQ3f * Z;
  o[14] = C15f * X * Z;
  o[15] = C15f * X * Y;
  o[16] = C5f * (Y * Y - 0.5f * (X * X + Z * Z));
  o[17] = C15f * Y * Z;
  o[18] = 0.5f * C15f * (Z * Z - X * X);
}

// ---------------------------------------------------------------------------
// o3_linear + norm_act for one of q/k/v. Input part layout:
//   part0: in[c]              c<m0   (d=1)
//   part1: in[p1off + c*3+m]  c<m1   (d=3)
//   part2: in[p2off + c*5+m]  c<m2   (d=5)
// Output 99 floats: [0..10], [11+i*3+m], [44+i*5+m]  (11 channels each part)
// ---------------------------------------------------------------------------
__device__ void o3_one(const float* __restrict__ in, int p1off, int p2off,
                       int m0, int m1, int m2,
                       const float* __restrict__ w0, const float* __restrict__ w1,
                       const float* __restrict__ w2, float* __restrict__ out) {
  float i0 = rsqrtf((float)m0), i1 = rsqrtf((float)m1), i2 = rsqrtf((float)m2);
  for (int i = 0; i < 11; ++i) {
    float a = 0.f;
    for (int c = 0; c < m0; ++c) a += in[c] * w0[c * 11 + i];
    a *= i0;
    out[i] = a * sigm(sqrtf(a * a + EPS2f));
  }
  for (int i = 0; i < 11; ++i) {
    float o[3], ss = 0.f;
    for (int m = 0; m < 3; ++m) {
      float a = 0.f;
      for (int c = 0; c < m1; ++c) a += in[p1off + c * 3 + m] * w1[c * 11 + i];
      a *= i1; o[m] = a; ss += a * a;
    }
    float sg = sigm(sqrtf(ss + EPS2f));
    for (int m = 0; m < 3; ++m) out[11 + i * 3 + m] = o[m] * sg;
  }
  for (int i = 0; i < 11; ++i) {
    float o[5], ss = 0.f;
    for (int m = 0; m < 5; ++m) {
      float a = 0.f;
      for (int c = 0; c < m2; ++c) a += in[p2off + c * 5 + m] * w2[c * 11 + i];
      a *= i2; o[m] = a; ss += a * a;
    }
    float sg = sigm(sqrtf(ss + EPS2f));
    for (int m = 0; m < 5; ++m) out[44 + i * 5 + m] = o[m] * sg;
  }
}

__global__ void k_qkv(const float* __restrict__ In, int istride, int p1off, int p2off,
                      int m0, int m1, int m2,
                      const float* q0, const float* q1, const float* q2,
                      const float* k0, const float* k1, const float* k2,
                      const float* v0, const float* v1, const float* v2,
                      float* __restrict__ Q, float* __restrict__ K,
                      float* __restrict__ V, int n) {
  int node = blockIdx.x * blockDim.x + threadIdx.x;
  if (node >= n) return;
  const float* in = In + (size_t)node * istride;
  o3_one(in, p1off, p2off, m0, m1, m2, q0, q1, q2, Q + (size_t)node * 99);
  o3_one(in, p1off, p2off, m0, m1, m2, k0, k1, k2, K + (size_t)node * 99);
  o3_one(in, p1off, p2off, m0, m1, m2, v0, v1, v2, V + (size_t)node * 99);
}

// ---------------------------------------------------------------------------
// K2: edge attention score: s = sum_parts einsum(q_i,k_j,w_ij)/sqrt(d) / sqrt(363)
// ex = exp(s); z[dst] += ex  (softmax denominator, atomics)
// ---------------------------------------------------------------------------
__global__ void k_edge(const float* __restrict__ Q, const float* __restrict__ K,
                       const int* __restrict__ src, const int* __restrict__ dst,
                       const float* __restrict__ w0, const float* __restrict__ w1,
                       const float* __restrict__ w2,
                       float* __restrict__ ex, float* __restrict__ z, int ne) {
  int e = blockIdx.x * blockDim.x + threadIdx.x;
  if (e >= ne) return;
  const float* q = Q + (size_t)dst[e] * 99;
  const float* k = K + (size_t)src[e] * 99;
  float a0 = 0.f, a1 = 0.f, a2 = 0.f;
  for (int i = 0; i < 11; ++i) {
    float qi = q[i];
    for (int j = 0; j < 11; ++j) {
      float w = w0[i * 11 + j];
      a0 += w * qi * k[j];
      float d3 = 0.f;
      for (int m = 0; m < 3; ++m) d3 += q[11 + i * 3 + m] * k[11 + j * 3 + m];
      a1 += w1[i * 11 + j] * d3;
      float d5 = 0.f;
      for (int m = 0; m < 5; ++m) d5 += q[44 + i * 5 + m] * k[44 + j * 5 + m];
      a2 += w2[i * 11 + j] * d5;
    }
  }
  float s = (a0 + a1 * INV_SQ3f + a2 * INV_SQ5f) * INV_SQ363f;
  float ev = __expf(s);
  ex[e] = ev;
  atomicAdd(&z[dst[e]], ev);
}

// ---------------------------------------------------------------------------
// K3: AGG[dst[e]] += sqrt(relu(ex/z[dst])) * V[src[e]]   (one thread per (e,j))
// ---------------------------------------------------------------------------
__global__ void k_agg(const float* __restrict__ ex, const float* __restrict__ z,
                      const float* __restrict__ V, const int* __restrict__ src,
                      const int* __restrict__ dst, float* __restrict__ AGG,
                      int ne) {
  long long idx = (long long)blockIdx.x * blockDim.x + threadIdx.x;
  if (idx >= (long long)ne * 99) return;
  int e = (int)(idx / 99), j = (int)(idx % 99);
  int d = dst[e];
  float zr = z[d]; if (zr == 0.f) zr = 1.0f;
  float w = sqrtf(fmaxf(ex[e] / zr, 0.f));
  atomicAdd(&AGG[(size_t)d * 99 + j], w * V[(size_t)src[e] * 99 + j]);
}

// K4: parts = (residual? old_parts : 0) + AGG + V[src[n]]   (N==E trick)
__global__ void k_finalize(float* __restrict__ P, const float* __restrict__ AGG,
                           const float* __restrict__ V, const int* __restrict__ src,
                           int residual, int n) {
  long long idx = (long long)blockIdx.x * blockDim.x + threadIdx.x;
  if (idx >= (long long)n * 99) return;
  int node = (int)(idx / 99), j = (int)(idx % 99);
  float val = AGG[idx] + V[(size_t)src[node] * 99 + j];
  P[idx] = residual ? (P[idx] + val) : val;
}

// ---------------------------------------------------------------------------
// K5: S = parts0 @ ol_w / sqrt(11), row L2-normalize, store f16 row-major.
// One wave per node (wave32); sum of squares via __shfl_xor.
// ---------------------------------------------------------------------------
__global__ void __launch_bounds__(256) k_head(const float* __restrict__ P,
                                              const float* __restrict__ olw,
                                              _Float16* __restrict__ S, int n) {
  int node = blockIdx.x * 8 + (threadIdx.x >> 5);
  int lane = threadIdx.x & 31;
  if (node >= n) return;
  float p0[11];
  for (int i = 0; i < 11; ++i) p0[i] = P[(size_t)node * 99 + i];
  float vals[8], ss = 0.f;
  for (int u = 0; u < 8; ++u) {
    int c = lane + 32 * u;
    float a = 0.f;
    for (int i = 0; i < 11; ++i) a += p0[i] * olw[i * 256 + c];
    a *= INV_SQ11f;
    vals[u] = a; ss += a * a;
  }
  for (int m = 16; m >= 1; m >>= 1) ss += __shfl_xor(ss, m, 32);
  float scale = 1.0f / fmaxf(sqrtf(ss), 1e-12f);
  for (int u = 0; u < 8; ++u)
    S[(size_t)node * 256 + lane + 32 * u] = (_Float16)(vals[u] * scale);
}

// ---------------------------------------------------------------------------
// K6a: pre-convert a 256x256 f32 weight into f16 laid out in B-fragment order:
//   WH[((nt*8+kt)*32+lane)*16+e] = W[(kt*32 + (lane>>4)*16 + e)*256 + nt*16 + (lane&15)]
// One thread per output element (65536).
// ---------------------------------------------------------------------------
__global__ void k_prepw(const float* __restrict__ W, _Float16* __restrict__ WH) {
  int idx = blockIdx.x * blockDim.x + threadIdx.x;
  if (idx >= 16 * 8 * 32 * 16) return;
  int e    = idx & 15;
  int lane = (idx >> 4) & 31;
  int kt   = (idx >> 9) & 7;
  int nt   = idx >> 12;
  int k = kt * 32 + ((lane >> 4) << 4) + e;
  int n = nt * 16 + (lane & 15);
  WH[idx] = (_Float16)W[k * 256 + n];
}

// ---------------------------------------------------------------------------
// K6: C[nrows,256] = relu(A[nrows,256] @ W + b), f16 in/out, f32 accumulate.
// Wave handles MT=5 row tiles x 1 col tile; B fragment (v16h, contiguous from
// the pre-swizzled WH) is reused across the 5 row tiles. A fragment = two
// contiguous v8h (b128) loads from row-major f16.
// A/B use contiguous k order k = 16*(lane>>4) + e (not the ISA table order); f32 C as in ISA 7.12.2.
// Grid: (nrows/80, 4), block 128 (4 waves).
// ---------------------------------------------------------------------------
__global__ void __launch_bounds__(128)
k_gemm256_relu(const _Float16* __restrict__ A, const _Float16* __restrict__ WH,
               const float* __restrict__ bias, _Float16* __restrict__ C, int nrows) {
  const int lane = threadIdx.x & 31;
  const int wave = threadIdx.x >> 5;
  const int tileN = blockIdx.y * 4 + wave;
  const int m0 = blockIdx.x * (16 * MT);
  const int n0 = tileN * 16;
  const int lm = lane & 15, kh = lane >> 4;

  v8f acc[MT] = {};
  const _Float16* arow[MT];
#pragma unroll
  for (int t = 0; t < MT; ++t) arow[t] = A + (size_t)(m0 + t * 16 + lm) * 256;
  const _Float16* wfrag = WH + ((size_t)tileN * 8 * 32 + lane) * 16;

#pragma unroll
  for (int kt8 = 0; kt8 < 8; ++kt8) {
    v16h bf = *(const v16h*)(wfrag + (size_t)kt8 * 32 * 16);
    const int ko = kt8 * 32 + kh * 16;
    if (kt8 < 7) __builtin_prefetch(arow[0] + ko + 32, 0, 1);  // global_prefetch_b8
#pragma unroll
    for (int t = 0; t < MT; ++t) {
      v8h lo = *(const v8h*)(arow[t] + ko);
      v8h hi = *(const v8h*)(arow[t] + ko + 8);
      v16h af;
#pragma unroll
      for (int e = 0; e < 8; ++e) { af[e] = lo[e]; af[8 + e] = hi[e]; }
      acc[t] = __builtin_amdgcn_wmma_f32_16x16x32_f16(false, af, false, bf,
                                                      (short)0, acc[t], false, false);
    }
  }
  const float b = bias[n0 + lm];
#pragma unroll
  for (int t = 0; t < MT; ++t) {
#pragma unroll
    for (int r = 0; r < 8; ++r) {
      int m = m0 + t * 16 + r + 8 * kh;
      float v = acc[t][r] + b;
      C[(size_t)m * 256 + (n0 + lm)] = (_Float16)(v > 0.f ? v : 0.f);
    }
  }
}

// K7: group sums: gsum[batch[n]] += S[dst[n]]; gcnt[batch[n]] += 1 (c==0 lane)
__global__ void k_groupsum(const _Float16* __restrict__ S, const int* __restrict__ dst,
                           const int* __restrict__ batch, float* __restrict__ gsum,
                           float* __restrict__ gcnt, int ne) {
  long long idx = (long long)blockIdx.x * blockDim.x + threadIdx.x;
  if (idx >= (long long)ne * 256) return;
  int n = (int)(idx >> 8), c = (int)(idx & 255);
  int g = batch[n];
  atomicAdd(&gsum[(size_t)g * 256 + c], (float)S[(size_t)dst[n] * 256 + c]);
  if (c == 0) atomicAdd(&gcnt[g], 1.0f);
}

// K8: m = gsum/max(cnt,1); out = softmax(m @ out_w + out_b)
__global__ void k_final(const float* __restrict__ gsum, const float* __restrict__ gcnt,
                        const float* __restrict__ ow, const float* __restrict__ ob,
                        float* __restrict__ out, int G) {
  int g = blockIdx.x * blockDim.x + threadIdx.x;
  if (g >= G) return;
  float cinv = 1.0f / fmaxf(gcnt[g], 1.0f);
  float logit[9];
  for (int o = 0; o < 9; ++o) logit[o] = ob[o];
  const float* gs = gsum + (size_t)g * 256;
  for (int c = 0; c < 256; ++c) {
    float m = gs[c] * cinv;
    for (int o = 0; o < 9; ++o) logit[o] += m * ow[c * 9 + o];
  }
  float mx = logit[0];
  for (int o = 1; o < 9; ++o) mx = fmaxf(mx, logit[o]);
  float se = 0.f;
  for (int o = 0; o < 9; ++o) { logit[o] = __expf(logit[o] - mx); se += logit[o]; }
  float inv = 1.0f / se;
  for (int o = 0; o < 9; ++o) out[(size_t)g * 9 + o] = logit[o] * inv;
}

// ---------------------------------------------------------------------------
// Host launch. Input order = setup_inputs() dict order, params flattened as a
// JAX pytree (dict keys alphabetical):
//  0 x, 1 pos, 2 edge_src, 3 edge_dst, 4 batch_ids,
//  5 embd_b, 6 embd_w,
//  layers[0]: 7-9 dot[0..2], 10-12 k[0..2], 13-15 q[0..2], 16-18 v[0..2]
//  layers[1]: 19-21 dot, 22-24 k, 25-27 q, 28-30 v
//  31 lin0_w, 32 lin0_b, 33 lin1_w, 34 lin1_b, 35 ol_w, 36 out_b, 37 out_w
// ---------------------------------------------------------------------------
extern "C" void kernel_launch(void* const* d_in, const int* in_sizes, int n_in,
                              void* d_out, int out_size, void* d_ws, size_t ws_size,
                              hipStream_t stream) {
  const float* x    = (const float*)d_in[0];
  const float* pos  = (const float*)d_in[1];
  const int*   esrc = (const int*)d_in[2];
  const int*   edst = (const int*)d_in[3];
  const int*   bid  = (const int*)d_in[4];
  const float* eb   = (const float*)d_in[5];
  const float* ew   = (const float*)d_in[6];
  const float* L0d[3] = {(const float*)d_in[7],  (const float*)d_in[8],  (const float*)d_in[9]};
  const float* L0k[3] = {(const float*)d_in[10], (const float*)d_in[11], (const float*)d_in[12]};
  const float* L0q[3] = {(const float*)d_in[13], (const float*)d_in[14], (const float*)d_in[15]};
  const float* L0v[3] = {(const float*)d_in[16], (const float*)d_in[17], (const float*)d_in[18]};
  const float* L1d[3] = {(const float*)d_in[19], (const float*)d_in[20], (const float*)d_in[21]};
  const float* L1k[3] = {(const float*)d_in[22], (const float*)d_in[23], (const float*)d_in[24]};
  const float* L1q[3] = {(const float*)d_in[25], (const float*)d_in[26], (const float*)d_in[27]};
  const float* L1v[3] = {(const float*)d_in[28], (const float*)d_in[29], (const float*)d_in[30]};
  const float* lin0w = (const float*)d_in[31];
  const float* lin0b = (const float*)d_in[32];
  const float* lin1w = (const float*)d_in[33];
  const float* lin1b = (const float*)d_in[34];
  const float* olw   = (const float*)d_in[35];
  const float* outb  = (const float*)d_in[36];
  const float* outw  = (const float*)d_in[37];

  const int N = in_sizes[1] / 3;   // 250000
  const int E = in_sizes[2];       // 250000
  const int G = out_size / 9;      // 500

  // workspace carving (each region aligned to 64B)
  float* ws  = (float*)d_ws;
  auto align16 = [](size_t v) { return (v + 15) & ~(size_t)15; };
  size_t off = 0;
  float* Pin = ws + off; off = align16(off + (size_t)N * 19);
  float* Qb  = ws + off; off = align16(off + (size_t)N * 99);
  float* Kb  = ws + off; off = align16(off + (size_t)N * 99);
  float* Vb  = ws + off; off = align16(off + (size_t)N * 99);
  float* P0  = ws + off; off = align16(off + (size_t)N * 99);
  float* AGG = ws + off; off = align16(off + (size_t)N * 99);
  float* EXb = ws + off; off = align16(off + (size_t)E);
  float* Zb  = ws + off; off = align16(off + (size_t)N);
  float* GS  = ws + off; off = align16(off + (size_t)G * 256);
  float* GC  = ws + off; off = align16(off + (size_t)G);
  _Float16* WH0 = (_Float16*)(ws + off); off = align16(off + (size_t)65536 / 2);
  _Float16* WH1 = (_Float16*)(ws + off); off = align16(off + (size_t)65536 / 2);
  _Float16* S0h = (_Float16*)(ws + off); off = align16(off + (size_t)N * 128);
  _Float16* S1h = (_Float16*)(ws + off);

  const int T = 256;
  dim3 bN((N + T - 1) / T), bE((E + T - 1) / T);
  dim3 b99(((long long)N * 99 + T - 1) / T);

  k_embed<<<bN, T, 0, stream>>>(x, pos, ew, eb, Pin, N);
  k_prepw<<<dim3(65536 / T), T, 0, stream>>>(lin0w, WH0);
  k_prepw<<<dim3(65536 / T), T, 0, stream>>>(lin1w, WH1);

  // ---- transformer layer 0 (muls_in = 11,1,1; input stride 19)
  k_qkv<<<bN, T, 0, stream>>>(Pin, 19, 11, 14, 11, 1, 1,
                              L0q[0], L0q[1], L0q[2],
                              L0k[0], L0k[1], L0k[2],
                              L0v[0], L0v[1], L0v[2], Qb, Kb, Vb, N);
  hipMemsetAsync(Zb, 0, (size_t)N * 4, stream);
  k_edge<<<bE, T, 0, stream>>>(Qb, Kb, esrc, edst, L0d[0], L0d[1], L0d[2], EXb, Zb, E);
  hipMemsetAsync(AGG, 0, (size_t)N * 99 * 4, stream);
  k_agg<<<dim3(((long long)E * 99 + T - 1) / T), T, 0, stream>>>(EXb, Zb, Vb, esrc, edst, AGG, E);
  k_finalize<<<b99, T, 0, stream>>>(P0, AGG, Vb, esrc, 0, N);

  // ---- transformer layer 1 (muls_in = 11,11,11; input stride 99) + residual
  k_qkv<<<bN, T, 0, stream>>>(P0, 99, 11, 44, 11, 11, 11,
                              L1q[0], L1q[1], L1q[2],
                              L1k[0], L1k[1], L1k[2],
                              L1v[0], L1v[1], L1v[2], Qb, Kb, Vb, N);
  hipMemsetAsync(Zb, 0, (size_t)N * 4, stream);
  k_edge<<<bE, T, 0, stream>>>(Qb, Kb, esrc, edst, L1d[0], L1d[1], L1d[2], EXb, Zb, E);
  hipMemsetAsync(AGG, 0, (size_t)N * 99 * 4, stream);
  k_agg<<<dim3(((long long)E * 99 + T - 1) / T), T, 0, stream>>>(EXb, Zb, Vb, esrc, edst, AGG, E);
  k_finalize<<<b99, T, 0, stream>>>(P0, AGG, Vb, esrc, 1, N);

  // ---- output head: 11->256 + row-normalize (f16 out), then 2x WMMA GEMM+ReLU
  k_head<<<dim3((N + 7) / 8), 256, 0, stream>>>(P0, olw, S0h, N);
  k_gemm256_relu<<<dim3(N / (16 * MT), 4), 128, 0, stream>>>(S0h, WH0, lin0b, S1h, N);
  k_gemm256_relu<<<dim3(N / (16 * MT), 4), 128, 0, stream>>>(S1h, WH1, lin1b, S0h, N);

  // ---- graph pooling + softmax
  hipMemsetAsync(GS, 0, (size_t)G * 256 * 4, stream);
  hipMemsetAsync(GC, 0, (size_t)G * 4, stream);
  k_groupsum<<<dim3(((long long)E * 256 + T - 1) / T), T, 0, stream>>>(S0h, edst, bid, GS, GC, E);
  k_final<<<dim3((G + 127) / 128), 128, 0, stream>>>(GS, GC, outw, outb, (float*)d_out, G);
}
